// STGLSTM_layer_44126493999324
// MI455X (gfx1250) — matrix operand order repaired and checked by reading, not compiled, not run on hardware
//
#include <hip/hip_runtime.h>
#include <hip/hip_bf16.h>
#include <math.h>

// ---------------------------------------------------------------------------
// STGLSTM layer for MI455X (gfx1250, wave32, WMMA bf16 16x16x32, f32 accum)
//   B=32, C=64, V=1024, T=24, C_OUT=64, GC=3C=192, gates=4C=256
// ---------------------------------------------------------------------------

typedef __bf16 bf16;
typedef __attribute__((ext_vector_type(16))) __bf16 v16bf;
typedef __attribute__((ext_vector_type(8)))  __bf16 v8bf;
typedef __attribute__((ext_vector_type(8)))  float   v8f;

#define BB    32
#define CC    64
#define VV    1024
#define TSTEPS 24
#define GC    192     // 3*C
#define G4    256     // 4*C
#define MROWS 2048    // B*C

// ---- scalar bf16 <-> f32 (round-to-nearest-even), no reliance on __bf16 arith
__device__ __forceinline__ bf16 f2bf(float f) {
  unsigned u = __builtin_bit_cast(unsigned, f);
  unsigned r = (u + 0x7FFFu + ((u >> 16) & 1u)) >> 16;
  unsigned short s = (unsigned short)r;
  return __builtin_bit_cast(bf16, s);
}
__device__ __forceinline__ float bf2f(bf16 b) {
  unsigned short s = __builtin_bit_cast(unsigned short, b);
  unsigned u = ((unsigned)s) << 16;
  return __builtin_bit_cast(float, u);
}

// ---- WMMA operand fragment loaders (layouts per CDNA5 ISA 7.12.2) ----------
// A-matrix 16x32 bf16: lanes 0-15 (M=0..15) hold K {0..7, 16..23};
//                      lanes 16-31 (M=0..15) hold K {8..15, 24..31}
__device__ __forceinline__ v16bf load_a16(const bf16* base, int stride,
                                          int m, int k, int lane) {
  int half = (lane >> 4) & 1;
  int row  = m + (lane & 15);
  const bf16* p = base + (size_t)row * stride + k + half * 8;
  v8bf lo = *(const v8bf*)p;          // K = k + half*8 + [0..7]
  v8bf hi = *(const v8bf*)(p + 16);   // K = k + 16 + half*8 + [0..7]
  return __builtin_shufflevector(lo, hi, 0,1,2,3,4,5,6,7,8,9,10,11,12,13,14,15);
}
// B-matrix 32x16 bf16 (element (k,n) read from base[n*stride + k]):
// lanes 0-15: N=lane, K {0..7, 16..23} ; lanes 16-31: N=lane-16, K {8..15, 24..31}
__device__ __forceinline__ v16bf load_b16(const bf16* base, int stride,
                                          int n, int k, int lane) {
  int half = (lane >> 4) & 1;
  int row  = n + (lane & 15);
  const bf16* p = base + (size_t)row * stride + k + half * 8;
  v8bf lo = *(const v8bf*)p;
  v8bf hi = *(const v8bf*)(p + 16);
  return __builtin_shufflevector(lo, hi, 0,1,2,3,4,5,6,7,8,9,10,11,12,13,14,15);
}
__device__ __forceinline__ v8f wmma_bf16(v16bf a, v16bf b, v8f c) {
  return __builtin_amdgcn_wmma_f32_16x16x32_bf16(false, a, false, b,
                                                 (short)0, c, false, false);
}

// ---------------------------------------------------------------------------
// Prep kernels
// ---------------------------------------------------------------------------

// x (B,C,V,T) f32  ->  xT (T, B*C, V) bf16   (LDS tile transpose, coalesced)
__global__ __launch_bounds__(256) void xpose_kernel(const float* __restrict__ x,
                                                    bf16* __restrict__ xT) {
  __shared__ bf16 tile[64][26];
  int bc = blockIdx.x;
  int v0 = blockIdx.y * 64;
  for (int u = threadIdx.x; u < 64 * TSTEPS; u += 256) {
    int vl = u / TSTEPS, tt = u % TSTEPS;
    tile[vl][tt] = f2bf(x[((size_t)bc * VV + v0 + vl) * TSTEPS + tt]);
  }
  __syncthreads();
  for (int u = threadIdx.x; u < 64 * TSTEPS; u += 256) {
    int tt = u >> 6, vl = u & 63;
    xT[((size_t)tt * MROWS + bc) * VV + v0 + vl] = tile[vl][tt];
  }
}

// A (V,V) f32 -> AT (V,V) bf16 with AT[w][v] = A[v][w]
__global__ __launch_bounds__(256) void atrans_kernel(const float* __restrict__ A,
                                                     bf16* __restrict__ AT) {
  __shared__ float tile[32][33];
  int v0 = blockIdx.x * 32, w0 = blockIdx.y * 32;
  int tx = threadIdx.x & 31, ty = threadIdx.x >> 5;
#pragma unroll
  for (int i = 0; i < 4; ++i) {
    int r = ty + i * 8;
    tile[r][tx] = A[(size_t)(v0 + r) * VV + w0 + tx];
  }
  __syncthreads();
#pragma unroll
  for (int i = 0; i < 4; ++i) {
    int r = ty + i * 8;
    AT[(size_t)(w0 + r) * VV + v0 + tx] = f2bf(tile[tx][r]);
  }
}

__global__ __launch_bounds__(256) void cvt_w_kernel(const float* __restrict__ Wg,
                                                    const float* __restrict__ Wo,
                                                    bf16* __restrict__ Wgb,
                                                    bf16* __restrict__ Wob) {
  int u = blockIdx.x * 256 + threadIdx.x;
  if (u < G4 * GC) Wgb[u] = f2bf(Wg[u]);
  if (u < CC * CC) Wob[u] = f2bf(Wo[u]);
}

__global__ __launch_bounds__(256) void zero_state_kernel(bf16* __restrict__ H,
                                                         float* __restrict__ Cst) {
  size_t u = (size_t)blockIdx.x * 256 + threadIdx.x;
  if (u < (size_t)MROWS * VV) { H[u] = f2bf(0.f); Cst[u] = 0.f; }
}

// ---------------------------------------------------------------------------
// Per-timestep kernels
// ---------------------------------------------------------------------------

// comb = x_t + h ; writes row-major comb bf16 and g-buffer slot 0 (channel-major)
__global__ __launch_bounds__(256) void comb_kernel(const bf16* __restrict__ xTt,
                                                   const bf16* __restrict__ H,
                                                   bf16* __restrict__ comb,
                                                   bf16* __restrict__ G) {
  int b  = blockIdx.x;
  int v  = blockIdx.y * 32 + (threadIdx.x & 31);
  int c0 = (threadIdx.x >> 5) * 8;
  v8bf p;
#pragma unroll
  for (int i = 0; i < 8; ++i) {
    size_t idx = ((size_t)b * CC + c0 + i) * VV + v;
    bf16 sb = f2bf(bf2f(xTt[idx]) + bf2f(H[idx]));
    comb[idx] = sb;
    p[i] = sb;
  }
  *(v8bf*)(G + ((size_t)b * VV + v) * GC + c0) = p;   // component 0
}

// Y = X @ A  (X row-major (2048,1024) bf16, A via AT).
// Block = 64M x 128N tile, 8 waves each 32x32 (2x2 WMMA tiles, 4 accumulators:
// both A-frags and both B-frags reused twice -> ~16 FLOP/B from cache).
// Writes row-major Y (optional) + g-buffer component `comp` (channel-major).
__global__ __launch_bounds__(256) void hop_kernel(const bf16* __restrict__ X,
                                                  const bf16* __restrict__ AT,
                                                  bf16* __restrict__ Y,
                                                  bf16* __restrict__ G, int comp) {
  __shared__ __align__(16) bf16 tile[64][136];   // pitch 136*2B = 272B (16B mult)
  int tid = threadIdx.x;
  int wave = tid >> 5, lane = tid & 31;
  int half = lane >> 4, ln = lane & 15;
  int wy = wave >> 2, wx = wave & 3;             // 2 (M) x 4 (N) wave grid
  int m0 = blockIdx.x * 64;
  int n0 = blockIdx.y * 128;
  int mb = m0 + wy * 32;                         // this wave: rows mb..mb+31
  int nb = n0 + wx * 32;                         // this wave: cols nb..nb+31

  v8f acc00 = {0,0,0,0,0,0,0,0};
  v8f acc01 = acc00, acc10 = acc00, acc11 = acc00;
  for (int kk = 0; kk < VV; kk += 32) {
    v16bf a0 = load_a16(X,  VV, mb,      kk, lane);
    v16bf a1 = load_a16(X,  VV, mb + 16, kk, lane);
    v16bf b0 = load_b16(AT, VV, nb,      kk, lane);
    v16bf b1 = load_b16(AT, VV, nb + 16, kk, lane);
    acc00 = wmma_bf16(a0, b0, acc00);
    acc01 = wmma_bf16(a0, b1, acc01);
    acc10 = wmma_bf16(a1, b0, acc10);
    acc11 = wmma_bf16(a1, b1, acc11);
  }
  int mloc = wy * 32 + half * 8;                 // + j ; second m-tile at +16
  int nloc = wx * 32 + ln;                       // second n-tile at +16
#pragma unroll
  for (int j = 0; j < 8; ++j) {
    tile[mloc + j     ][nloc     ] = f2bf(acc00[j]);
    tile[mloc + j     ][nloc + 16] = f2bf(acc01[j]);
    tile[mloc + j + 16][nloc     ] = f2bf(acc10[j]);
    tile[mloc + j + 16][nloc + 16] = f2bf(acc11[j]);
  }
  __syncthreads();

  if (Y) {  // row-major bf16 (feeds next hop's A-operand), coalesced 16B stores
    for (int u = tid; u < 1024; u += 256) {
      int r = u >> 4, s = (u & 15) * 8;
      *(v8bf*)(Y + (size_t)(m0 + r) * VV + n0 + s) = *(const v8bf*)&tile[r][s];
    }
  }
  // channel-major write into g: G[(b*V+v)*GC + comp*64 + c], b == blockIdx.x
  int b = blockIdx.x;
  for (int u = tid; u < 1024; u += 256) {
    int vloc = u >> 3, ch = (u & 7) * 8;
    v8bf p;
#pragma unroll
    for (int i = 0; i < 8; ++i) p[i] = tile[ch + i][vloc];
    *(v8bf*)(G + ((size_t)b * VV + n0 + vloc) * GC + comp * 64 + ch) = p;
  }
}

// gates = Wg @ g_b (+bias), LSTM pointwise, out_t = Wo @ hy (+bias).
// One block per (b, 16-wide v tile); 8 waves cover the 256 gate rows.
__global__ __launch_bounds__(256) void cell_kernel(const bf16* __restrict__ G,
                                                   const bf16* __restrict__ Wg,
                                                   const float* __restrict__ bg,
                                                   const bf16* __restrict__ Wo,
                                                   const float* __restrict__ bo,
                                                   bf16* __restrict__ H,
                                                   float* __restrict__ Cst,
                                                   float* __restrict__ out, int t) {
  __shared__ float gl[G4][17];
  __shared__ __align__(16) bf16 hyl[16][64];   // [v][c] for projection B-operand
  int b  = blockIdx.x;
  int v0 = blockIdx.y * 16;
  int tid = threadIdx.x;
  int wave = tid >> 5, lane = tid & 31;
  int half = lane >> 4, ln = lane & 15;

  const bf16* Gb = G + (size_t)b * VV * GC;
  int m0 = wave * 32, m1 = m0 + 16;
  v8f acc0 = {0,0,0,0,0,0,0,0};
  v8f acc1 = acc0;
  for (int kk = 0; kk < GC; kk += 32) {
    v16bf bfrag = load_b16(Gb, GC, v0, kk, lane);
    v16bf a0 = load_a16(Wg, GC, m0, kk, lane);
    v16bf a1 = load_a16(Wg, GC, m1, kk, lane);
    acc0 = wmma_bf16(a0, bfrag, acc0);
    acc1 = wmma_bf16(a1, bfrag, acc1);
  }
#pragma unroll
  for (int j = 0; j < 8; ++j) {
    int r0 = m0 + half * 8 + j, r1 = m1 + half * 8 + j;
    gl[r0][ln] = acc0[j] + bg[r0];
    gl[r1][ln] = acc1[j] + bg[r1];
  }
  __syncthreads();

  for (int u = tid; u < CC * 16; u += 256) {
    int cch = u >> 4, vl = u & 15;
    float ig = gl[cch][vl];
    float fg = gl[64  + cch][vl];
    float gg = gl[128 + cch][vl];
    float og = gl[192 + cch][vl];
    size_t cidx = ((size_t)b * CC + cch) * VV + v0 + vl;
    float si = 1.f / (1.f + __expf(-ig));
    float sf = 1.f / (1.f + __expf(-fg));
    float so = 1.f / (1.f + __expf(-og));
    float cy = sf * Cst[cidx] + si * gg;       // note: cell gate NOT tanh'd (ref)
    Cst[cidx] = cy;
    float hy = so * tanhf(cy);
    bf16 hb = f2bf(hy);
    H[cidx] = hb;                               // next step's h (row-major bf16)
    hyl[vl][cch] = hb;
  }
  __syncthreads();

  if (wave < 4) {                               // out = Wo(64x64) @ hy(64x16)
    int mo = wave * 16;
    v8f acc = {0,0,0,0,0,0,0,0};
    const bf16* hb = &hyl[0][0];
    for (int kk = 0; kk < CC; kk += 32) {
      v16bf a  = load_a16(Wo, CC, mo, kk, lane);
      v16bf bf_ = load_b16(hb, CC, 0, kk, lane);   // ds_load_b128 from LDS
      acc = wmma_bf16(a, bf_, acc);
    }
#pragma unroll
    for (int j = 0; j < 8; ++j) {
      int o = mo + half * 8 + j;
      out[(((size_t)b * CC + o) * VV + v0 + ln) * TSTEPS + t] = acc[j] + bo[o];
    }
  }
}

// ---------------------------------------------------------------------------
extern "C" void kernel_launch(void* const* d_in, const int* in_sizes, int n_in,
                              void* d_out, int out_size, void* d_ws, size_t ws_size,
                              hipStream_t stream) {
  (void)in_sizes; (void)n_in; (void)out_size; (void)ws_size;
  const float* x  = (const float*)d_in[0];
  const float* A  = (const float*)d_in[1];
  const float* Wg = (const float*)d_in[2];
  const float* bg = (const float*)d_in[3];
  const float* Wo = (const float*)d_in[4];
  const float* bo = (const float*)d_in[5];
  float* out = (float*)d_out;

  char* w = (char*)d_ws;
  size_t off = 0;
  auto take = [&](size_t bytes) -> char* {
    char* p = w + off;
    off += (bytes + 255) & ~((size_t)255);
    return p;
  };
  bf16* xT   = (bf16*)take((size_t)TSTEPS * MROWS * VV * 2);  // ~100.7 MB
  bf16* AT   = (bf16*)take((size_t)VV * VV * 2);              // 2 MB (L2-hot)
  bf16* Wgb  = (bf16*)take((size_t)G4 * GC * 2);
  bf16* Wob  = (bf16*)take((size_t)CC * CC * 2);
  bf16* comb = (bf16*)take((size_t)MROWS * VV * 2);
  bf16* x1   = (bf16*)take((size_t)MROWS * VV * 2);
  bf16* G    = (bf16*)take((size_t)BB * VV * GC * 2);         // (B,V,3C)
  bf16* H    = (bf16*)take((size_t)MROWS * VV * 2);
  float* Cst = (float*)take((size_t)MROWS * VV * 4);

  // one-time prep
  xpose_kernel<<<dim3(MROWS, VV / 64), 256, 0, stream>>>(x, xT);
  atrans_kernel<<<dim3(VV / 32, VV / 32), 256, 0, stream>>>(A, AT);
  cvt_w_kernel<<<dim3((G4 * GC + 255) / 256), 256, 0, stream>>>(Wg, Wo, Wgb, Wob);
  zero_state_kernel<<<dim3((MROWS * VV + 255) / 256), 256, 0, stream>>>(H, Cst);

  // sequential scan over T (launch order provides the inter-kernel syncs)
  for (int t = 0; t < TSTEPS; ++t) {
    const bf16* xTt = xT + (size_t)t * MROWS * VV;
    comb_kernel<<<dim3(BB, VV / 32), 256, 0, stream>>>(xTt, H, comb, G);
    hop_kernel<<<dim3(MROWS / 64, VV / 128), 256, 0, stream>>>(comb, AT, x1, G, 1);
    hop_kernel<<<dim3(MROWS / 64, VV / 128), 256, 0, stream>>>(x1, AT, nullptr, G, 2);
    cell_kernel<<<dim3(BB, VV / 16), 256, 0, stream>>>(G, Wgb, bg, Wob, bo,
                                                       H, Cst, out, t);
  }
}
